// MultiHeadAttention_48017734370052
// MI455X (gfx1250) — matrix operand order repaired and checked by reading, not compiled, not run on hardware
//
#include <hip/hip_runtime.h>
#include <hip/hip_bf16.h>
#include <math.h>

typedef __attribute__((ext_vector_type(16))) _Float16 v16h;
typedef __attribute__((ext_vector_type(8)))  _Float16 v8h;
typedef __attribute__((ext_vector_type(8)))  float    v8f;

#define WMMA_F16(a, b, c) \
  __builtin_amdgcn_wmma_f32_16x16x32_f16(false, (a), false, (b), (short)0, (c), false, false)

static constexpr int Bb = 2, S = 2048, E = 1024, H = 16, DK = 64;
static constexpr int M  = Bb * S;                 // 4096 rows
static constexpr int BM = 128, BN = 128, BK = 32; // GEMM block tile
static constexpr int LP = 40;                     // padded LDS row (halves): 80B stride

// async copy of 16 bytes global -> LDS (ASYNCcnt-tracked, CDNA5)
__device__ __forceinline__ void async_ld16(const _Float16* g, const _Float16* l) {
  unsigned lds_off = (unsigned)(unsigned long long)(const void*)l;
  asm volatile("global_load_async_to_lds_b128 %0, %1, off"
               :: "v"(lds_off), "v"(g) : "memory");
}

// ---------------------------------------------------------------- f32 -> f16
__global__ __launch_bounds__(256) void cvt_f32_f16(const float* __restrict__ in,
                                                   _Float16* __restrict__ out, int n) {
  int i = blockIdx.x * 256 + threadIdx.x;
  if (i < n) out[i] = (_Float16)in[i];
}

// ---------------------------------------------- async double-buffered WMMA GEMM
// C(M x E) = A(M x E, f16) @ W^T + bias, W stored (E x E) row-major.
// Block: 256 thr = 8 waves, 128x128 tile; wave = 64x32 (4x2 WMMA frags).
// mode 0/1: f16 Q/K as [b,h,s,d];  mode 2: f16 V^T as [b,h,d,s];  mode 3: f32 [m,n]
__global__ __launch_bounds__(256) void gemm_wmma_async(const _Float16* __restrict__ A,
                                                       const _Float16* __restrict__ W,
                                                       const float* __restrict__ bias,
                                                       void* __restrict__ outp, int mode) {
  __shared__ _Float16 At[2][BM][LP];   // 20 KB
  __shared__ _Float16 Bt[2][BN][LP];   // 20 KB

  int lane = threadIdx.x & 31;
  int wid  = threadIdx.x >> 5;
  int l15  = lane & 15;
  int h2   = lane >> 4;

  int bm  = blockIdx.x >> 3;           // M / 128 = 32
  int bn  = blockIdx.x & 7;            // E / 128 = 8
  int gm0 = bm * BM, gn0 = bn * BN;

  int wr = wid >> 2;                   // wave row (0..1) -> 64 rows
  int wc = wid & 3;                    // wave col (0..3) -> 32 cols

  int crow = lane >> 2;                // copy mapping: 8 rows / instr
  int cseg = lane & 3;                 // 4 x 16B segments per 32-half row

  auto load_stage = [&](int kk, int buf) {
    int k0 = kk * BK;
#pragma unroll
    for (int t = 0; t < 2; ++t) {      // 2 A + 2 B copies per wave per stage
      int row = wid * 16 + t * 8 + crow;
      async_ld16(A + (size_t)(gm0 + row) * E + k0 + cseg * 8, &At[buf][row][cseg * 8]);
      async_ld16(W + (size_t)(gn0 + row) * E + k0 + cseg * 8, &Bt[buf][row][cseg * 8]);
    }
  };

  v8f acc[4][2] = {};

  load_stage(0, 0);
  const int NK = E / BK;               // 32 k-steps
  for (int kk = 0; kk < NK; ++kk) {
    int buf = kk & 1;
    if (kk + 1 < NK) {
      load_stage(kk + 1, buf ^ 1);     // prefetch next stage (overlaps compute)
      asm volatile("s_wait_asynccnt 0x4" ::: "memory");  // oldest 4 (stage kk) done
    } else {
      asm volatile("s_wait_asynccnt 0x0" ::: "memory");
    }
    __syncthreads();                   // stage kk visible to all 8 waves

    v16h am[4];
#pragma unroll
    for (int i = 0; i < 4; ++i) {      // A fragments from LDS
      const _Float16* ar = &At[buf][wr * 64 + i * 16 + l15][0];
      union { v16h v; v8h h[2]; } u;
      u.h[0] = *(const v8h*)(ar + h2 * 16);
      u.h[1] = *(const v8h*)(ar + h2 * 16 + 8);
      am[i] = u.v;
    }
#pragma unroll
    for (int j = 0; j < 2; ++j) {      // B fragments + 8 WMMAs
      const _Float16* br = &Bt[buf][wc * 32 + j * 16 + l15][0];
      v16h bf = *(const v16h*)(br + h2 * 16);
#pragma unroll
      for (int i = 0; i < 4; ++i)
        acc[i][j] = WMMA_F16(am[i], bf, acc[i][j]);
    }
    __syncthreads();                   // everyone done reading buf before reuse
  }

  // ----- epilogue -----
#pragma unroll
  for (int i = 0; i < 4; ++i) {
    int gmb = gm0 + wr * 64 + i * 16;
#pragma unroll
    for (int j = 0; j < 2; ++j) {
      int gn = gn0 + wc * 32 + j * 16 + l15;
      float bs = bias[gn];
      if (mode == 3) {
        float* out = (float*)outp;
#pragma unroll
        for (int r = 0; r < 8; ++r) {
          int gm = gmb + r + 8 * h2;
          out[(size_t)gm * E + gn] = acc[i][j][r] + bs;
        }
      } else {
        _Float16* out = (_Float16*)outp;
        int h = gn >> 6, d = gn & (DK - 1);
#pragma unroll
        for (int r = 0; r < 8; ++r) {
          int gm = gmb + r + 8 * h2;
          int b  = gm >> 11;           // S = 2048
          int s  = gm & (S - 1);
          size_t addr;
          if (mode == 2) addr = ((size_t)(b * H + h) * DK + d) * S + s;  // V^T
          else           addr = ((size_t)(b * H + h) * S + s) * DK + d;  // Q / K
          out[addr] = (_Float16)(acc[i][j][r] + bs);
        }
      }
    }
  }
}

// ------------------------------------------------------ interleaved RoPE (in place)
__global__ __launch_bounds__(256) void rope_inplace(_Float16* __restrict__ buf) {
  int i = blockIdx.x * 256 + threadIdx.x;      // Bb*H*S*32 threads
  int p = i & 31;
  int s = (i >> 5) & (S - 1);
  size_t base = ((size_t)(i >> 5)) * DK + 2 * p;
  float invf  = exp2f(-(float)(2 * p) * (13.287712379549449f / 64.0f)); // 10000^(-2p/64)
  float theta = (float)s * invf;
  float cs = cosf(theta), sn = sinf(theta);
  float u0 = (float)buf[base], u1 = (float)buf[base + 1];
  buf[base]     = (_Float16)(u0 * cs - u1 * sn);
  buf[base + 1] = (_Float16)(u1 * cs + u0 * sn);
}

// ------------------------------------------------------------ flash attention
// Q,K: [B*H, S, 64] f16 (RoPE applied).  Vt: [B*H, 64, S] f16.
// Ao:  [B, S, E] f16.  One wave = one 16-row Q tile, online softmax, causal.
__global__ __launch_bounds__(256) void flash_attn(const _Float16* __restrict__ Q,
                                                  const _Float16* __restrict__ K,
                                                  const _Float16* __restrict__ Vt,
                                                  _Float16* __restrict__ Ao) {
  __shared__ _Float16 plds[8][16][32];
  int lane = threadIdx.x & 31;
  int wid  = threadIdx.x >> 5;
  int l15  = lane & 15;
  int h2   = lane >> 4;

  int gid = blockIdx.x * 8 + wid;
  int qt  = gid & (S / 16 - 1);
  int bh  = gid >> 7;
  int q0  = qt * 16;

  const _Float16* Qb = Q  + (size_t)bh * S * DK;
  const _Float16* Kb = K  + (size_t)bh * S * DK;
  const _Float16* Vb = Vt + (size_t)bh * DK * S;

  v16h aq[2];
  {
    const _Float16* qr = Qb + (size_t)(q0 + l15) * DK;
#pragma unroll
    for (int kc = 0; kc < 2; ++kc) {
      union { v16h v; v8h h[2]; } u;
      u.h[0] = *(const v8h*)(qr + kc * 32 + h2 * 16);
      u.h[1] = *(const v8h*)(qr + kc * 32 + h2 * 16 + 8);
      aq[kc] = u.v;
    }
  }

  v8f vo[4] = {};
  float mrow[8], lrow[8];
#pragma unroll
  for (int r = 0; r < 8; ++r) { mrow[r] = -__builtin_inff(); lrow[r] = 0.0f; }

  const float SC = 0.125f * 1.44269504088896340736f; // 1/sqrt(64) * log2(e)

  int steps = (qt + 2) >> 1;
  for (int j = 0; j < steps; ++j) {
    int kv0 = j * 32;
    v8f sc[2];
#pragma unroll
    for (int t = 0; t < 2; ++t) {
      int kbase = kv0 + t * 16;
      if (kbase > q0) {
#pragma unroll
        for (int r = 0; r < 8; ++r) sc[t][r] = -__builtin_inff();
        continue;
      }
      const _Float16* kr = Kb + (size_t)(kbase + l15) * DK;
      v8f c = {};
#pragma unroll
      for (int kc = 0; kc < 2; ++kc) {
        v16h bk = *(const v16h*)(kr + kc * 32 + h2 * 16);
        c = WMMA_F16(aq[kc], bk, c);
      }
      int gk = kbase + l15;
#pragma unroll
      for (int r = 0; r < 8; ++r) {
        float v = c[r] * SC;
        if (kbase == q0 && gk > q0 + r + 8 * h2) v = -__builtin_inff();
        sc[t][r] = v;
      }
    }

#pragma unroll
    for (int r = 0; r < 8; ++r) {
      float mx = fmaxf(sc[0][r], sc[1][r]);
#pragma unroll
      for (int off = 8; off > 0; off >>= 1)
        mx = fmaxf(mx, __shfl_xor(mx, off, 32));
      float mnew  = fmaxf(mrow[r], mx);
      float p0    = exp2f(sc[0][r] - mnew);
      float p1    = exp2f(sc[1][r] - mnew);
      float alpha = exp2f(mrow[r] - mnew);
      float rs = p0 + p1;
#pragma unroll
      for (int off = 8; off > 0; off >>= 1)
        rs += __shfl_xor(rs, off, 32);
      lrow[r] = lrow[r] * alpha + rs;
      mrow[r] = mnew;
#pragma unroll
      for (int f = 0; f < 4; ++f) vo[f][r] *= alpha;
      plds[wid][r + 8 * h2][l15]      = (_Float16)p0;
      plds[wid][r + 8 * h2][16 + l15] = (_Float16)p1;
    }

    asm volatile("s_wait_dscnt 0" ::: "memory");  // same-wave LDS turnaround

    union { v16h v; v8h h[2]; } up;
    up.h[0] = *(const v8h*)(&plds[wid][l15][h2 * 16]);
    up.h[1] = *(const v8h*)(&plds[wid][l15][h2 * 16 + 8]);
#pragma unroll
    for (int f = 0; f < 4; ++f) {
      v16h bv = *(const v16h*)(Vb + (size_t)(f * 16 + l15) * S + kv0 + h2 * 16);
      vo[f] = WMMA_F16(up.v, bv, vo[f]);
    }
  }

  int b = bh >> 4, h = bh & (H - 1);
#pragma unroll
  for (int r = 0; r < 8; ++r) {
    float inv = 1.0f / lrow[r];
    int s = q0 + r + 8 * h2;
    size_t base = ((size_t)(b * S + s)) * E + h * DK;
#pragma unroll
    for (int f = 0; f < 4; ++f)
      Ao[base + f * 16 + l15] = (_Float16)(vo[f][r] * inv);
  }
}

// ---------------------------------------------------------------------- launch
extern "C" void kernel_launch(void* const* d_in, const int* in_sizes, int n_in,
                              void* d_out, int out_size, void* d_ws, size_t ws_size,
                              hipStream_t stream) {
  (void)in_sizes; (void)n_in; (void)out_size; (void)ws_size;
  const float* x    = (const float*)d_in[0];
  const float* wq_w = (const float*)d_in[1];
  const float* wq_b = (const float*)d_in[2];
  const float* wk_w = (const float*)d_in[3];
  const float* wk_b = (const float*)d_in[4];
  const float* wv_w = (const float*)d_in[5];
  const float* wv_b = (const float*)d_in[6];
  const float* wo_w = (const float*)d_in[7];
  const float* wo_b = (const float*)d_in[8];
  float* out = (float*)d_out;

  char* p = (char*)d_ws;
  _Float16* Xh  = (_Float16*)p; p += (size_t)M * E * 2;
  _Float16* Wqh = (_Float16*)p; p += (size_t)E * E * 2;
  _Float16* Wkh = (_Float16*)p; p += (size_t)E * E * 2;
  _Float16* Wvh = (_Float16*)p; p += (size_t)E * E * 2;
  _Float16* Woh = (_Float16*)p; p += (size_t)E * E * 2;
  _Float16* Qb  = (_Float16*)p; p += (size_t)M * E * 2;   // [B*H, S, 64]
  _Float16* Kb  = (_Float16*)p; p += (size_t)M * E * 2;   // [B*H, S, 64]
  _Float16* Vtb = (_Float16*)p; p += (size_t)M * E * 2;   // [B*H, 64, S]
  _Float16* Ao  = (_Float16*)p; p += (size_t)M * E * 2;   // [B, S, E]

  cvt_f32_f16<<<(M * E + 255) / 256, 256, 0, stream>>>(x,    Xh,  M * E);
  cvt_f32_f16<<<(E * E + 255) / 256, 256, 0, stream>>>(wq_w, Wqh, E * E);
  cvt_f32_f16<<<(E * E + 255) / 256, 256, 0, stream>>>(wk_w, Wkh, E * E);
  cvt_f32_f16<<<(E * E + 255) / 256, 256, 0, stream>>>(wv_w, Wvh, E * E);
  cvt_f32_f16<<<(E * E + 255) / 256, 256, 0, stream>>>(wo_w, Woh, E * E);

  int gblocks = (M / BM) * (E / BN);       // 256 blocks, 8 waves each
  gemm_wmma_async<<<gblocks, 256, 0, stream>>>(Xh, Wqh, wq_b, (void*)Qb,  0);
  gemm_wmma_async<<<gblocks, 256, 0, stream>>>(Xh, Wkh, wk_b, (void*)Kb,  1);
  gemm_wmma_async<<<gblocks, 256, 0, stream>>>(Xh, Wvh, wv_b, (void*)Vtb, 2);

  int rblocks = (Bb * H * S * 32) / 256;
  rope_inplace<<<rblocks, 256, 0, stream>>>(Qb);
  rope_inplace<<<rblocks, 256, 0, stream>>>(Kb);

  int ablocks = (Bb * H * (S / 16)) / 8;
  flash_attn<<<ablocks, 256, 0, stream>>>(Qb, Kb, Vtb, Ao);

  gemm_wmma_async<<<gblocks, 256, 0, stream>>>(Ao, Woh, wo_b, (void*)out, 3);
}
